// GraphQNN_65481071397799
// MI455X (gfx1250) — matrix operand order repaired and checked by reading, not compiled, not run on hardware
//
#include <hip/hip_runtime.h>
#include <hip/hip_bf16.h>

// GraphQNN (graph attention) for MI455X / gfx1250, wave32 + WMMA f16->f32.
//
// Per layer l:
//   Qf16 = (h @ Wq + bq) * 1/sqrt(D)      (proj_kernel, f16 row-major out)
//   Kf16 =  h @ Wk + bk                   (proj_kernel, f16 row-major out)
//   Vt   = (h @ Wv + bv)^T                (proj_kernel, f16 transposed out)
//   Ha   = flash_attention(Qf16,Kf16,Vt, mask-bits)    (f16 out)
//   out[l+1] = Ha @ Wo + bo               (proj_kernel, f32 out -> d_out)
//
// Flash kernel stages K/V tiles into LDS with gfx1250 async global->LDS
// copies (ASYNCcnt), double-buffered so block i+1 fills while block i is
// consumed by WMMA: cuts WGP-ingress traffic 8x (K/V shared by 8 waves).

typedef __attribute__((ext_vector_type(16))) _Float16 v16h;
typedef __attribute__((ext_vector_type(8)))  _Float16 v8h;
typedef __attribute__((ext_vector_type(8)))  float    v8f;

#define NN 8192
#define DD 128
#define NEGBIG (-3.0e38f)

__device__ __forceinline__ v8f wmma_fp16(v16h a, v16h b, v8f c) {
  return __builtin_amdgcn_wmma_f32_16x16x32_f16(
      /*neg_a=*/false, a, /*neg_b=*/false, b,
      /*c_mod=*/(short)0, c, /*reuse_a=*/false, /*reuse_b=*/false);
}

// Generic LDS pointer = {shared_aperture_hi32, lds_offset32}; the async-load
// VDST operand wants the wave-relative LDS byte offset = low 32 bits.
__device__ __forceinline__ unsigned lds_off(const void* p) {
  return (unsigned)(size_t)p;
}

// GLOBAL_LOAD_ASYNC_TO_LDS_B128: per-lane 16B global->LDS, tracked by ASYNCcnt.
__device__ __forceinline__ void async_b128(unsigned ldsoff, const void* gaddr) {
  asm volatile("global_load_async_to_lds_b128 %0, %1, off"
               :: "v"(ldsoff), "v"(gaddr) : "memory");
}
__device__ __forceinline__ void wait_async0() {
  asm volatile("s_wait_asynccnt 0" ::: "memory");
}

// A-matrix fragment (16x32, f16): row m = lane&15, k = hi*16 + j
__device__ __forceinline__ v16h load_a_f16(const _Float16* base, int row,
                                           int kbase, int hi) {
  const _Float16* p = base + (size_t)row * DD + kbase + hi * 16;
  v8h lo = *(const v8h*)p;
  v8h hh = *(const v8h*)(p + 8);
  v16h a;
#pragma unroll
  for (int j = 0; j < 8; ++j) { a[j] = lo[j]; a[8 + j] = hh[j]; }
  return a;
}

__device__ __forceinline__ v16h load_a_f32(const float* base, int row,
                                           int kbase, int hi) {
  const float* p = base + (size_t)row * DD + kbase + hi * 16;
  v16h a;
#pragma unroll
  for (int j = 0; j < 8; ++j) {
    a[j]     = (_Float16)p[j];
    a[8 + j] = (_Float16)p[8 + j];
  }
  return a;
}

// ---------------------------------------------------------------------------
// Prep kernels
// ---------------------------------------------------------------------------
__global__ void repack_weights(const float* Wq, const float* Wk,
                               const float* Wv, const float* Wo,
                               _Float16* wfrag) {
  int idx = blockIdx.x * blockDim.x + threadIdx.x;
  if (idx >= 8 * DD * DD) return;
  int mat = idx / (DD * DD);
  int e   = idx % (DD * DD);
  int j    = e & 15;
  int lane = (e >> 4) & 31;
  int kc   = (e >> 9) & 3;
  int nt   = e >> 11;
  int hi = lane >> 4, nlo = lane & 15;
  int k = kc * 32 + hi * 16 + j;
  int n = nt * 16 + nlo;
  int l = mat >> 2, which = mat & 3;
  const float* W = (which == 0) ? Wq : (which == 1) ? Wk : (which == 2) ? Wv : Wo;
  wfrag[idx] = (_Float16)W[(size_t)l * DD * DD + (size_t)k * DD + n];
}

__global__ void pack_mask(const unsigned char* mask, unsigned int* bits) {
  int idx = blockIdx.x * blockDim.x + threadIdx.x;
  if (idx >= NN * NN / 32) return;
  const unsigned char* p = mask + (size_t)idx * 32;
  unsigned int w = 0;
#pragma unroll
  for (int i = 0; i < 32; ++i) w |= (p[i] ? 1u : 0u) << i;
  bits[idx] = w;
}

// ---------------------------------------------------------------------------
// Projection GEMM: [NN x DD] @ [DD x DD] + bias. One wave -> 16 rows x 128 cols.
// AMODE: 0 = f32 activations, 1 = f16. OMODE: 1 = f16 (scaled), 2 = f16^T, 3 = f32.
// ---------------------------------------------------------------------------
template <int AMODE, int OMODE>
__global__ __launch_bounds__(256) void proj_kernel(const void* __restrict__ hsrc,
                                                   const _Float16* __restrict__ wfrag,
                                                   const float* __restrict__ bias,
                                                   void* __restrict__ out,
                                                   float scale) {
  int lane = threadIdx.x & 31;
  int wave = threadIdx.x >> 5;
  int rowbase = (blockIdx.x * 8 + wave) * 16;
  int hi = lane >> 4, nlo = lane & 15;

  v16h a[4];
#pragma unroll
  for (int kc = 0; kc < 4; ++kc)
    a[kc] = (AMODE == 0)
                ? load_a_f32((const float*)hsrc, rowbase + nlo, kc * 32, hi)
                : load_a_f16((const _Float16*)hsrc, rowbase + nlo, kc * 32, hi);

#pragma unroll
  for (int nt = 0; nt < 8; ++nt) {
    v8f acc = {};
#pragma unroll
    for (int kc = 0; kc < 4; ++kc) {
      v16h b = *(const v16h*)(wfrag + (size_t)(((nt * 4 + kc) * 32) + lane) * 16);
      acc = wmma_fp16(a[kc], b, acc);
    }
    float bcol = bias[nt * 16 + nlo];
    int col = nt * 16 + nlo;
#pragma unroll
    for (int r = 0; r < 8; ++r) {
      int row = rowbase + r + 8 * hi;
      float v = (acc[r] + bcol) * scale;
      if (OMODE == 3)
        ((float*)out)[(size_t)row * DD + col] = v;
      else if (OMODE == 2)
        ((_Float16*)out)[(size_t)col * NN + row] = (_Float16)v;  // V^T
      else
        ((_Float16*)out)[(size_t)row * DD + col] = (_Float16)v;
    }
  }
}

// ---------------------------------------------------------------------------
// Flash attention with async double-buffered K/V staging in LDS.
// Block = 8 waves, each wave owns a 16-row tile; 32-key inner blocks.
// ---------------------------------------------------------------------------
#define KPAD 136  // 32-key K rows: 128 + 8 halves  (272B, 16B aligned)
#define VPAD 40   // V^T rows: 32 + 8 halves        (80B, 16B aligned)

__device__ __forceinline__ void stage_kv(const _Float16* __restrict__ Kf,
                                         const _Float16* __restrict__ Vt,
                                         int kb, int tid,
                                         _Float16 (*Kb)[KPAD],
                                         _Float16 (*Vb)[VPAD]) {
  // K block: 32 nodes x 128 feats = 512 x 16B chunks; V: 128 feats x 32 keys.
#pragma unroll
  for (int i = 0; i < 2; ++i) {
    int c = tid + i * 256;
    int node = c >> 4, part = c & 15;
    async_b128(lds_off(&Kb[node][part * 8]),
               Kf + (size_t)(kb + node) * DD + part * 8);
  }
#pragma unroll
  for (int i = 0; i < 2; ++i) {
    int c = tid + i * 256;
    int feat = c >> 2, part = c & 3;
    async_b128(lds_off(&Vb[feat][part * 8]),
               Vt + (size_t)feat * NN + kb + part * 8);
  }
}

__global__ __launch_bounds__(256) void flash_kernel(
    const _Float16* __restrict__ Qf, const _Float16* __restrict__ Kf,
    const _Float16* __restrict__ Vt, const unsigned int* __restrict__ mbits,
    _Float16* __restrict__ hattn) {
  __shared__ _Float16 Kbuf[2][32][KPAD];   // 17 KB
  __shared__ _Float16 Vbuf[2][DD][VPAD];   // 20 KB
  __shared__ _Float16 pbuf[8][16][40];     // 10 KB, per-wave P transpose

  int tid  = threadIdx.x;
  int lane = tid & 31;
  int wave = tid >> 5;
  int rowbase = (blockIdx.x * 8 + wave) * 16;
  int hi = lane >> 4, nlo = lane & 15;

  v16h q[4];
#pragma unroll
  for (int kc = 0; kc < 4; ++kc)
    q[kc] = load_a_f16(Qf, rowbase + nlo, kc * 32, hi);

  v8f zero = {};
  v8f o[8];
#pragma unroll
  for (int t = 0; t < 8; ++t) o[t] = zero;
  float mrun[8], lrun[8];
#pragma unroll
  for (int r = 0; r < 8; ++r) { mrun[r] = NEGBIG; lrun[r] = 0.0f; }

  // Prime buffer 0.
  stage_kv(Kf, Vt, 0, tid, Kbuf[0], Vbuf[0]);
  wait_async0();
  __syncthreads();

  for (int kb = 0; kb < NN; kb += 32) {
    int cur = (kb >> 5) & 1;
    if (kb + 32 < NN)  // prefetch next block into the other buffer
      stage_kv(Kf, Vt, kb + 32, tid, Kbuf[cur ^ 1], Vbuf[cur ^ 1]);

    // ---- S = Q @ K^T for two 16-col tiles (8 WMMAs, B from LDS) ----
    v8f s0 = zero, s1 = zero;
#pragma unroll
    for (int kc = 0; kc < 4; ++kc) {
      v16h b0 = *(const v16h*)&Kbuf[cur][nlo][kc * 32 + hi * 16];
      s0 = wmma_fp16(q[kc], b0, s0);
    }
#pragma unroll
    for (int kc = 0; kc < 4; ++kc) {
      v16h b1 = *(const v16h*)&Kbuf[cur][16 + nlo][kc * 32 + hi * 16];
      s1 = wmma_fp16(q[kc], b1, s1);
    }

    // ---- bit-mask + online softmax; write P tile to per-wave LDS ----
#pragma unroll
    for (int r = 0; r < 8; ++r) {
      int m = r + 8 * hi;  // row within tile (C layout)
      int row = rowbase + m;
      unsigned int w = mbits[(size_t)row * (NN / 32) + (kb >> 5)];
      float v0 = ((w >> nlo) & 1u) ? s0[r] : NEGBIG;
      float v1 = ((w >> (16 + nlo)) & 1u) ? s1[r] : NEGBIG;

      float mt = fmaxf(v0, v1);
#pragma unroll
      for (int d = 1; d < 16; d <<= 1) mt = fmaxf(mt, __shfl_xor(mt, d, 32));
      float mnew = fmaxf(mrun[r], mt);
      float corr = __expf(mrun[r] - mnew);
      float p0 = __expf(v0 - mnew);
      float p1 = __expf(v1 - mnew);
      float rs = p0 + p1;
#pragma unroll
      for (int d = 1; d < 16; d <<= 1) rs += __shfl_xor(rs, d, 32);
      lrun[r] = lrun[r] * corr + rs;
      mrun[r] = mnew;
#pragma unroll
      for (int t = 0; t < 8; ++t) o[t][r] *= corr;
      pbuf[wave][m][nlo]      = (_Float16)p0;
      pbuf[wave][m][16 + nlo] = (_Float16)p1;
    }

    // ---- P back in A layout (same-wave DS ordering, no barrier) ----
    const _Float16* pr = &pbuf[wave][nlo][hi * 16];
    v8h plo = *(const v8h*)pr;
    v8h phi = *(const v8h*)(pr + 8);
    v16h pa;
#pragma unroll
    for (int j = 0; j < 8; ++j) { pa[j] = plo[j]; pa[8 + j] = phi[j]; }

    // ---- O += P @ V  (8 WMMAs, B from staged V^T in LDS) ----
#pragma unroll
    for (int t = 0; t < 8; ++t) {
      v16h bv = *(const v16h*)&Vbuf[cur][t * 16 + nlo][hi * 16];
      o[t] = wmma_fp16(pa, bv, o[t]);
    }

    // All ds reads were consumed by WMMAs above; fence our async fills and
    // rendezvous: after this barrier buf^1 is ready, buf cur is reusable.
    wait_async0();
    __syncthreads();
  }

  // ---- epilogue: normalize, store f16 for the output projection ----
#pragma unroll
  for (int r = 0; r < 8; ++r) {
    float inv = 1.0f / lrun[r];
    size_t rowoff = (size_t)(rowbase + r + 8 * hi) * DD;
#pragma unroll
    for (int t = 0; t < 8; ++t)
      hattn[rowoff + t * 16 + nlo] = (_Float16)(o[t][r] * inv);
  }
}

// ---------------------------------------------------------------------------
extern "C" void kernel_launch(void* const* d_in, const int* in_sizes, int n_in,
                              void* d_out, int out_size, void* d_ws, size_t ws_size,
                              hipStream_t stream) {
  const float*         features = (const float*)d_in[0];
  const unsigned char* mask     = (const unsigned char*)d_in[1];  // bool, 1B/elt
  const float* Wq = (const float*)d_in[2];
  const float* bq = (const float*)d_in[3];
  const float* Wk = (const float*)d_in[4];
  const float* bk = (const float*)d_in[5];
  const float* Wv = (const float*)d_in[6];
  const float* bv = (const float*)d_in[7];
  const float* Wo = (const float*)d_in[8];
  const float* bo = (const float*)d_in[9];

  float* out = (float*)d_out;
  char* ws = (char*)d_ws;

  _Float16*     wfrag = (_Float16*)ws;                              // 256 KB
  unsigned int* mbits = (unsigned int*)(ws + (256 << 10));          // 8 MB
  _Float16*     Qf    = (_Float16*)(ws + (256 << 10) + (8 << 20));  // 2 MB
  _Float16*     Kf    = Qf + (size_t)NN * DD;                       // 2 MB
  _Float16*     Vt    = Kf + (size_t)NN * DD;                       // 2 MB
  _Float16*     Ha    = Vt + (size_t)NN * DD;                       // 2 MB

  repack_weights<<<(8 * DD * DD + 255) / 256, 256, 0, stream>>>(Wq, Wk, Wv, Wo, wfrag);
  pack_mask<<<(NN * NN / 32 + 255) / 256, 256, 0, stream>>>(mask, mbits);

  hipMemcpyAsync(out, features, (size_t)NN * DD * sizeof(float),
                 hipMemcpyDeviceToDevice, stream);  // outputs[0] = features

  const float scale = 0.08838834764831845f;  // 1/sqrt(128)
  const size_t MS = (size_t)DD * DD;

  for (int l = 0; l < 2; ++l) {
    const float* h = out + (size_t)l * NN * DD;
    proj_kernel<0, 1><<<64, 256, 0, stream>>>(h, wfrag + (l * 4 + 0) * MS,
                                              bq + (size_t)l * DD, Qf, scale);
    proj_kernel<0, 1><<<64, 256, 0, stream>>>(h, wfrag + (l * 4 + 1) * MS,
                                              bk + (size_t)l * DD, Kf, 1.0f);
    proj_kernel<0, 2><<<64, 256, 0, stream>>>(h, wfrag + (l * 4 + 2) * MS,
                                              bv + (size_t)l * DD, Vt, 1.0f);
    flash_kernel<<<64, 256, 0, stream>>>(Qf, Kf, Vt, mbits, Ha);
    proj_kernel<1, 3><<<64, 256, 0, stream>>>(Ha, wfrag + (l * 4 + 3) * MS,
                                              bo + (size_t)l * DD,
                                              out + (size_t)(l + 1) * NN * DD, 1.0f);
  }
}
